// NeuronCircuit_9990093931272
// MI455X (gfx1250) — matrix operand order repaired and checked by reading, not compiled, not run on hardware
//
#include <hip/hip_runtime.h>
#include <hip/hip_bf16.h>
#include <cstdint>

typedef __attribute__((ext_vector_type(16))) _Float16 v16h;
typedef __attribute__((ext_vector_type(8)))  _Float16 v8h;
typedef __attribute__((ext_vector_type(8)))  float    v8f;
typedef __attribute__((ext_vector_type(4)))  float    vf4;

#define NTOK   16384   // B*S = 4*4096
#define DMODEL 1024
#define RANK   256
#define NEXP   16      // N_IN == N_OUT == 16
#define KHH    8
#define LDSK   64      // K-chunk (halves) staged per LDS buffer
#define LDSP   72      // padded LDS row stride in halves (144B = 36 banks -> conflict-free)
#define MAXT64 272     // NTOK/64 + NEXP padding tiles (64-row tiles)

// meta layout (ints): cnt_in[16] @0, cnt_out[16] @16, off_in[17] @32,
// off_out[17] @49, tb_in[17] @66 (64-row tiles), tb_out[17] @83,
// cur_in[16] @100, cur_out[16] @116
#define M_CNT_IN  0
#define M_CNT_OUT 16
#define M_OFF_IN  32
#define M_OFF_OUT 49
#define M_TB_IN   66
#define M_TB_OUT  83
#define M_CUR_IN  100
#define M_CUR_OUT 116

// ---- prep: transpose + f32->f16 convert weights ----
// Wt_in[e][r][d] = W_in[e][d][r]   (B operand for input GEMM: contiguous in d)
__global__ void k_cvt_in(const float* __restrict__ W, _Float16* __restrict__ Wt) {
  long i = (long)blockIdx.x * 256 + threadIdx.x;           // 16*256*1024 = 4M
  int d = (int)(i & 1023);
  int r = (int)((i >> 10) & 255);
  int e = (int)(i >> 18);
  Wt[i] = (_Float16)W[((long)e * DMODEL + d) * RANK + r];
}
// Wt_out[e][d][r] = W_out[e][r][d] (B operand for output GEMM: contiguous in r)
__global__ void k_cvt_out(const float* __restrict__ W, _Float16* __restrict__ Wt) {
  long i = (long)blockIdx.x * 256 + threadIdx.x;           // 16*1024*256 = 4M
  int r = (int)(i & 255);
  int d = (int)((i >> 8) & 1023);
  int e = (int)(i >> 18);
  Wt[i] = (_Float16)W[((long)e * RANK + r) * DMODEL + d];
}

// ---- binning ----
__global__ void k_zero(int* meta) { if (threadIdx.x < 160) meta[threadIdx.x] = 0; }

__global__ void k_count(const int* __restrict__ inIdx, const int* __restrict__ outIdx,
                        int* __restrict__ meta) {
  int t = blockIdx.x * 256 + threadIdx.x;
  if (t < NTOK) {
    atomicAdd(&meta[M_CNT_IN  + inIdx[t]], 1);
    atomicAdd(&meta[M_CNT_OUT + outIdx[t]], 1);
  }
}

__global__ void k_scan(int* meta) {
  if (threadIdx.x == 0) {
    int a = 0, tt = 0;
    for (int e = 0; e < NEXP; e++) {
      meta[M_OFF_IN + e] = a; meta[M_TB_IN + e] = tt; meta[M_CUR_IN + e] = a;
      int c = meta[M_CNT_IN + e]; a += c; tt += (c + 63) >> 6;
    }
    meta[M_OFF_IN + 16] = a; meta[M_TB_IN + 16] = tt;
    a = 0; tt = 0;
    for (int e = 0; e < NEXP; e++) {
      meta[M_OFF_OUT + e] = a; meta[M_TB_OUT + e] = tt; meta[M_CUR_OUT + e] = a;
      int c = meta[M_CNT_OUT + e]; a += c; tt += (c + 63) >> 6;
    }
    meta[M_OFF_OUT + 16] = a; meta[M_TB_OUT + 16] = tt;
  }
}

__global__ void k_scatter(const int* __restrict__ inIdx, const int* __restrict__ outIdx,
                          int* __restrict__ meta, int* __restrict__ permIn,
                          int* __restrict__ permOut) {
  int t = blockIdx.x * 256 + threadIdx.x;
  if (t < NTOK) {
    int pi = atomicAdd(&meta[M_CUR_IN  + inIdx[t]], 1);  permIn[pi]  = t;
    int po = atomicAdd(&meta[M_CUR_OUT + outIdx[t]], 1); permOut[po] = t;
  }
}

// async stage of one 16-byte element into LDS
__device__ __forceinline__ void async_b128(unsigned lds_addr, const _Float16* g) {
  asm volatile("global_load_async_to_lds_b128 %0, %1, off"
               :: "v"(lds_addr), "v"(g) : "memory");
}
__device__ __forceinline__ void wait_async0() {
  asm volatile("s_wait_asynccnt 0x0" ::: "memory");
}
// read one B fragment (halves kc..kc+7 and kc+16..kc+23) from padded LDS row
__device__ __forceinline__ v16h lds_bfrag(const _Float16* lp, int row, int kc) {
  const _Float16* rp = lp + row * LDSP + kc;   // 16B-aligned (144B stride)
  v8h p0 = *(const v8h*)(rp);
  v8h p1 = *(const v8h*)(rp + 16);
  return __builtin_shufflevector(p0, p1, 0, 1, 2, 3, 4, 5, 6, 7,
                                 8, 9, 10, 11, 12, 13, 14, 15);
}

// ---- input GEMM: h[t][r] = sum_d x[t][d] * Win[e][d][r], grouped by expert ----
// block = 4 waves, tile = 64 tokens x 64 r. B panel staged to LDS (async ping-pong),
// K=1024 in 64-chunks -> 2 substeps x 4 x v_wmma_f32_16x16x32_f16 per chunk.
__global__ __launch_bounds__(128) void k_gemm_in(const float* __restrict__ x,
    const _Float16* __restrict__ Wt, const int* __restrict__ perm,
    const int* __restrict__ meta, float* __restrict__ hbuf) {
  __shared__ _Float16 ldsB[2][64 * LDSP];   // [row r(64)][k(64)+pad], 9 KB each
  const int* cnt = meta + M_CNT_IN;
  const int* off = meta + M_OFF_IN;
  const int* tb  = meta + M_TB_IN;
  int tileIdx = blockIdx.x;
  if (tileIdx >= tb[16]) return;                 // uniform exit
  int e = 0;
  #pragma unroll
  for (int i = 0; i < NEXP; i++) if (tileIdx >= tb[i + 1]) e = i + 1;

  int tid  = threadIdx.x;
  int lane = tid & 31;
  int w    = tid >> 5;
  int mL = lane & 15;
  int hi = lane >> 4;
  int cntE = cnt[e];
  int segBase = off[e];
  int rowStart = (tileIdx - tb[e]) * 64 + w * 16;

  int rr = rowStart + mL; if (rr >= cntE) rr = cntE - 1;   // pad rows duplicate last
  int tokA = perm[segBase + rr];
  const float* xrow = x + (long)tokA * DMODEL;

  int r0 = blockIdx.y * 64;
  const _Float16* wb = Wt + ((long)e * RANK + r0) * DMODEL;   // panel base
  unsigned lbase0 = (unsigned)(uintptr_t)(&ldsB[0][0]);
  unsigned lbase1 = (unsigned)(uintptr_t)(&ldsB[1][0]);

  // stage chunk [kb, kb+64) of 64 rows into buffer: 512 x 16B, 4 per thread
  auto stage = [&](int kb, unsigned lb) {
    #pragma unroll
    for (int it = 0; it < 4; it++) {
      int j = tid + 128 * it;          // 0..511
      int row  = j >> 3;               // 8 x 16B per row
      int colb = (j & 7) * 16;         // byte offset in row
      async_b128(lb + row * (LDSP * 2) + colb,
                 wb + (long)row * DMODEL + kb + (colb >> 1));
    }
  };

  v8f acc0 = {}, acc1 = {}, acc2 = {}, acc3 = {};
  stage(0, lbase0);
  const int NCH = DMODEL / LDSK;       // 16 chunks
  for (int c = 0; c < NCH; c++) {
    const _Float16* lp = (c & 1) ? &ldsB[1][0] : &ldsB[0][0];
    wait_async0();                     // my async writes done
    __syncthreads();                   // everyone's writes done; prev buf reads done
    if (c + 1 < NCH) stage((c + 1) * LDSK, (c & 1) ? lbase0 : lbase1);
    #pragma unroll
    for (int s = 0; s < 2; s++) {
      int kb = c * LDSK + s * 32;
      // A fragment (ISA layout): lane<16: halves0-7=K0-7, 8-15=K16-23; lane>=16: +8
      const float* pa = xrow + kb + hi * 8;
      vf4 f0 = *(const vf4*)(pa);
      vf4 f1 = *(const vf4*)(pa + 4);
      vf4 f2 = *(const vf4*)(pa + 16);
      vf4 f3 = *(const vf4*)(pa + 20);
      v16h a;
      #pragma unroll
      for (int j = 0; j < 4; j++) {
        a[j]      = (_Float16)f0[j];
        a[4 + j]  = (_Float16)f1[j];
        a[8 + j]  = (_Float16)f2[j];
        a[12 + j] = (_Float16)f3[j];
      }
      int kc = s * 32 + hi * 8;
      v16h b0 = lds_bfrag(lp, mL,      kc);
      v16h b1 = lds_bfrag(lp, 16 + mL, kc);
      v16h b2 = lds_bfrag(lp, 32 + mL, kc);
      v16h b3 = lds_bfrag(lp, 48 + mL, kc);
      acc0 = __builtin_amdgcn_wmma_f32_16x16x32_f16(false, a, false, b0, (short)0, acc0, false, false);
      acc1 = __builtin_amdgcn_wmma_f32_16x16x32_f16(false, a, false, b1, (short)0, acc1, false, false);
      acc2 = __builtin_amdgcn_wmma_f32_16x16x32_f16(false, a, false, b2, (short)0, acc2, false, false);
      acc3 = __builtin_amdgcn_wmma_f32_16x16x32_f16(false, a, false, b3, (short)0, acc3, false, false);
    }
  }
  // C layout: lane<16 -> rows j, lane>=16 -> rows j+8; column = lane&15
  int nCol = r0 + mL;
  int mofs = hi * 8;
  #pragma unroll
  for (int j = 0; j < 8; j++) {
    int r = rowStart + mofs + j;
    if (r < cntE) {
      int t = perm[segBase + r];
      float* hp = hbuf + (long)t * RANK + nCol;
      hp[0] = acc0[j]; hp[16] = acc1[j]; hp[32] = acc2[j]; hp[48] = acc3[j];
    }
  }
}

// ---- Householder chain: 1 wave per token, wave32 shuffle reductions ----
__global__ __launch_bounds__(32) void k_house(const float* __restrict__ hbuf,
    const float* __restrict__ pv, const int* __restrict__ pIdx,
    _Float16* __restrict__ h16) {
  int t = blockIdx.x;
  int lane = threadIdx.x;
  float h[8];
  #pragma unroll
  for (int i = 0; i < 8; i++) h[i] = hbuf[(long)t * RANK + i * 32 + lane];
  const int* pk = pIdx + (long)t * KHH;
  for (int k = 0; k < KHH; k++) {
    const float* v = pv + (long)pk[k] * RANK;
    float vr[8]; float vv = 0.f, dv = 0.f;
    #pragma unroll
    for (int i = 0; i < 8; i++) {
      vr[i] = v[i * 32 + lane];
      vv += vr[i] * vr[i];
      dv += h[i] * vr[i];
    }
    #pragma unroll
    for (int o = 16; o > 0; o >>= 1) {
      vv += __shfl_xor(vv, o, 32);
      dv += __shfl_xor(dv, o, 32);
    }
    float s = 2.0f * dv / (vv + 1e-8f);   // == 2*(h . v_n)*v_n, v_n = v/sqrt(vv+eps)
    #pragma unroll
    for (int i = 0; i < 8; i++) h[i] -= s * vr[i];
  }
  #pragma unroll
  for (int i = 0; i < 8; i++) h16[(long)t * RANK + i * 32 + lane] = (_Float16)h[i];
}

// ---- output GEMM: out[t][d] = sum_r h[t][r] * Wout[e][r][d], grouped by expert ----
__global__ __launch_bounds__(128) void k_gemm_out(const _Float16* __restrict__ h16,
    const _Float16* __restrict__ Wt, const int* __restrict__ perm,
    const int* __restrict__ meta, float* __restrict__ out) {
  __shared__ _Float16 ldsB[2][64 * LDSP];
  const int* cnt = meta + M_CNT_OUT;
  const int* off = meta + M_OFF_OUT;
  const int* tb  = meta + M_TB_OUT;
  int tileIdx = blockIdx.x;
  if (tileIdx >= tb[16]) return;
  int e = 0;
  #pragma unroll
  for (int i = 0; i < NEXP; i++) if (tileIdx >= tb[i + 1]) e = i + 1;

  int tid  = threadIdx.x;
  int lane = tid & 31;
  int w    = tid >> 5;
  int mL = lane & 15;
  int hi = lane >> 4;
  int cntE = cnt[e];
  int segBase = off[e];
  int rowStart = (tileIdx - tb[e]) * 64 + w * 16;

  int rr = rowStart + mL; if (rr >= cntE) rr = cntE - 1;
  int tokA = perm[segBase + rr];
  const _Float16* hrow = h16 + (long)tokA * RANK;

  int d0 = blockIdx.y * 64;
  const _Float16* wb = Wt + ((long)e * DMODEL + d0) * RANK;
  unsigned lbase0 = (unsigned)(uintptr_t)(&ldsB[0][0]);
  unsigned lbase1 = (unsigned)(uintptr_t)(&ldsB[1][0]);

  auto stage = [&](int kb, unsigned lb) {
    #pragma unroll
    for (int it = 0; it < 4; it++) {
      int j = tid + 128 * it;
      int row  = j >> 3;
      int colb = (j & 7) * 16;
      async_b128(lb + row * (LDSP * 2) + colb,
                 wb + (long)row * RANK + kb + (colb >> 1));
    }
  };

  v8f acc0 = {}, acc1 = {}, acc2 = {}, acc3 = {};
  stage(0, lbase0);
  const int NCH = RANK / LDSK;        // 4 chunks
  for (int c = 0; c < NCH; c++) {
    const _Float16* lp = (c & 1) ? &ldsB[1][0] : &ldsB[0][0];
    wait_async0();
    __syncthreads();
    if (c + 1 < NCH) stage((c + 1) * LDSK, (c & 1) ? lbase0 : lbase1);
    #pragma unroll
    for (int s = 0; s < 2; s++) {
      int kb = c * LDSK + s * 32;
      v8h lo  = *(const v8h*)(hrow + kb + hi * 8);
      v8h hi8 = *(const v8h*)(hrow + kb + 16 + hi * 8);
      v16h a = __builtin_shufflevector(lo, hi8, 0, 1, 2, 3, 4, 5, 6, 7,
                                       8, 9, 10, 11, 12, 13, 14, 15);
      int kc = s * 32 + hi * 8;
      v16h b0 = lds_bfrag(lp, mL,      kc);
      v16h b1 = lds_bfrag(lp, 16 + mL, kc);
      v16h b2 = lds_bfrag(lp, 32 + mL, kc);
      v16h b3 = lds_bfrag(lp, 48 + mL, kc);
      acc0 = __builtin_amdgcn_wmma_f32_16x16x32_f16(false, a, false, b0, (short)0, acc0, false, false);
      acc1 = __builtin_amdgcn_wmma_f32_16x16x32_f16(false, a, false, b1, (short)0, acc1, false, false);
      acc2 = __builtin_amdgcn_wmma_f32_16x16x32_f16(false, a, false, b2, (short)0, acc2, false, false);
      acc3 = __builtin_amdgcn_wmma_f32_16x16x32_f16(false, a, false, b3, (short)0, acc3, false, false);
    }
  }
  int nCol = d0 + mL;
  int mofs = hi * 8;
  #pragma unroll
  for (int j = 0; j < 8; j++) {
    int r = rowStart + mofs + j;
    if (r < cntE) {
      int t = perm[segBase + r];
      float* op = out + (long)t * DMODEL + nCol;
      op[0] = acc0[j]; op[16] = acc1[j]; op[32] = acc2[j]; op[48] = acc3[j];
    }
  }
}

extern "C" void kernel_launch(void* const* d_in, const int* in_sizes, int n_in,
                              void* d_out, int out_size, void* d_ws, size_t ws_size,
                              hipStream_t stream) {
  (void)in_sizes; (void)n_in; (void)out_size; (void)ws_size;
  const float* x     = (const float*)d_in[0];   // [4,4096,1024]
  const float* Win   = (const float*)d_in[1];   // [16,1024,256]
  const float* Vproc = (const float*)d_in[2];   // [64,256]
  const float* Wout  = (const float*)d_in[3];   // [16,256,1024]
  const int* inIdx   = (const int*)d_in[4];     // [4,4096]
  const int* pIdx    = (const int*)d_in[5];     // [4,4096,8]
  const int* outIdx  = (const int*)d_in[6];     // [4,4096]
  float* out = (float*)d_out;

  char* ws = (char*)d_ws;
  _Float16* WtIn  = (_Float16*)(ws);                         // 8 MB
  _Float16* WtOut = (_Float16*)(ws + ((size_t)8  << 20));    // 8 MB
  float*    hbuf  = (float*)   (ws + ((size_t)16 << 20));    // 16 MB
  _Float16* h16   = (_Float16*)(ws + ((size_t)32 << 20));    // 8 MB
  int* permIn  = (int*)(ws + ((size_t)40 << 20));            // 64 KB
  int* permOut = (int*)(ws + ((size_t)40 << 20) + 65536);    // 64 KB
  int* meta    = (int*)(ws + ((size_t)40 << 20) + 131072);   // <1 KB

  k_cvt_in <<<16384, 256, 0, stream>>>(Win,  WtIn);
  k_cvt_out<<<16384, 256, 0, stream>>>(Wout, WtOut);
  k_zero   <<<1, 256, 0, stream>>>(meta);
  k_count  <<<NTOK / 256, 256, 0, stream>>>(inIdx, outIdx, meta);
  k_scan   <<<1, 1, 0, stream>>>(meta);
  k_scatter<<<NTOK / 256, 256, 0, stream>>>(inIdx, outIdx, meta, permIn, permOut);

  k_gemm_in <<<dim3(MAXT64, RANK / 64),   128, 0, stream>>>(x, WtIn, permIn, meta, hbuf);
  k_house   <<<NTOK,                      32,  0, stream>>>(hbuf, Vproc, pIdx, h16);
  k_gemm_out<<<dim3(MAXT64, DMODEL / 64), 128, 0, stream>>>(h16, WtOut, permOut, meta, out);
}
